// RandomlyWiredStage_11149735100795
// MI455X (gfx1250) — matrix operand order repaired and checked by reading, not compiled, not run on hardware
//
#include <hip/hip_runtime.h>

typedef _Float16 v16h  __attribute__((ext_vector_type(16)));
typedef float    v8f   __attribute__((ext_vector_type(8)));
typedef float    f32x4 __attribute__((ext_vector_type(4)));
typedef int      i32x4 __attribute__((vector_size(16)));   // matches builtin signature

#define NB   128   // batch
#define NCH  128   // CIN
#define NH   28
#define NW   28
#define PH   30    // padded spatial (1-halo)
#define PW   30
#define NEMB 64
#define EPS_CUT 0.01f
#define NN_EPS 1e-5f
#define NN_MOM 0.1f

#if defined(__gfx1250__) && __has_builtin(__builtin_amdgcn_global_load_async_to_lds_b128)
#define HAVE_ASYNC_LDS 1
#else
#define HAVE_ASYNC_LDS 0
#endif

// ---- static graph structure tables (device) ----
__constant__ int c_ntar[7]     = {3,3,2,3,2,1,1};
__constant__ int c_tar[7][3]   = {{1,2,3},{2,3,4},{4,5,0},{4,5,6},{5,6,0},{7,0,0},{7,0,0}};
__constant__ int c_first[7][3] = {{1,1,1},{0,0,1},{0,1,0},{0,0,1},{0,0,0},{1,0,0},{0,0,0}};
__constant__ int c_npart[7]    = {0,3,3,3,2,2,1};
__constant__ int c_part[7][3]  = {{0,0,0},{2,3,4},{3,4,5},{4,5,6},{5,6,0},{6,7,0},{7,0,0}};
__constant__ int c_nsrc[8]     = {0,1,2,2,3,3,2,2};
__constant__ int c_src[8][3]   = {{0,0,0},{0,0,0},{0,1,0},{0,1,0},{1,2,3},{2,3,4},{3,4,0},{5,6,0}};

// ---------------- init: zero attention state, copy running stats ----------------
__global__ void k_init(float* attn_dist, float* node_attns, float* sb, float* tm,
                       float* rm, float* rv, const float* run_mean, const float* run_var) {
  int t = threadIdx.x;
  for (int i = t; i < NB*8; i += blockDim.x) { attn_dist[i] = 0.f; node_attns[i] = 0.f; }
  for (int i = t; i < 8*NB;  i += blockDim.x) sb[i] = (i < NB) ? 1.f : 0.f;   // sb[0][b]=1
  for (int i = t; i < 64*NB; i += blockDim.x) tm[i] = 0.f;
  if (t < 8) { rm[t] = run_mean[t]; rv[t] = run_var[t]; }
}

// ---------------- zero the padded f16 NHWC input buffer (halo stays 0) ----------------
__global__ void k_zerop(_Float16* __restrict__ p, size_t n8) {
  size_t i = (size_t)blockIdx.x * blockDim.x + threadIdx.x;
  if (i >= n8) return;
  const f32x4 z = {0.f, 0.f, 0.f, 0.f};
  ((f32x4*)p)[i] = z;
}

// ---------------- weight pre-transpose to f16 co-major [Co][r*128+ci] ----------------
__global__ void k_wconv(const float* __restrict__ w, _Float16* __restrict__ wt, int Co) {
  int i = blockIdx.x * blockDim.x + threadIdx.x;
  int total = Co * NCH * 9;
  if (i >= total) return;
  int co  = i / (NCH*9);
  int rem = i % (NCH*9);
  int r   = rem / NCH;
  int ci  = rem % NCH;
  wt[i] = (_Float16)w[(co*NCH + ci)*9 + r];
}

// ---------------- global mean pool: pooled[b*C+c] = mean_{hw} ----------------
__global__ void k_pool(const float* __restrict__ x, float* __restrict__ pooled, int C, int HW) {
  int wid  = (blockIdx.x * blockDim.x + threadIdx.x) >> 5;
  int lane = threadIdx.x & 31;
  int rows = NB * C;
  if (wid >= rows) return;
  const float* p = x + (size_t)wid * HW;
  float s = 0.f;
  for (int i = lane; i < HW; i += 32) s += p[i];
  for (int off = 16; off > 0; off >>= 1) s += __shfl_down(s, off, 32);
  if (lane == 0) pooled[wid] = s / (float)HW;
}

// ---------------- block reduction helper ----------------
__device__ float block_sum(float v, float* red) {
  int b = threadIdx.x;
  __syncthreads();
  red[b] = v;
  __syncthreads();
  for (int s = 64; s > 0; s >>= 1) { if (b < s) red[b] += red[b + s]; __syncthreads(); }
  return red[0];
}

// ---------------- per-node attention / BN / softmax / top-k cut ----------------
__global__ void k_attn(int nid,
                       const float* __restrict__ pooled,
                       const float* __restrict__ attend_w,   // [128][64]
                       const float* __restrict__ node_embs,  // [8][64]
                       const float* __restrict__ node_gamma,
                       const float* __restrict__ node_beta,
                       float* rm, float* rv,
                       float* attn_dist, float* node_attns, float* sb, float* tm)
{
  __shared__ float red[NB];
  int b = threadIdx.x;                   // 128 threads, one per batch row
  int ntar = c_ntar[nid];
  float sbv = sb[nid*NB + b];
  float attn_sent[3] = {0.f, 0.f, 0.f};

  if (ntar > 1) {
    const float* pb = pooled + b*NCH;
    float scores[3];
    for (int j = 0; j < ntar; ++j) {
      int t = c_tar[nid][j];
      const float* emb = node_embs + t*NEMB;
      float s = 0.f;
      for (int e = 0; e < NEMB; ++e) {
        float qv = 0.f;
        for (int c = 0; c < NCH; ++c) qv += pb[c] * attend_w[c*NEMB + e];
        s += qv * emb[e];
      }
      scores[j] = s * 0.125f;
    }
    float cnt = fmaxf(block_sum(sbv, red), 1.f);
    float m = NN_MOM * cnt / (float)NB;
    float ns[3];
    for (int j = 0; j < ntar; ++j) {
      int t = c_tar[nid][j];
      float old_mean = rm[t], old_var = rv[t];
      float mu  = block_sum(scores[j] * sbv, red) / cnt;
      float dv  = scores[j] - old_mean;
      float var = block_sum(dv * dv * sbv, red) / cnt;
      float new_mean = old_mean * (1.f - m) + mu * m;
      float new_var  = old_var  * (1.f - m) + var * m;
      __syncthreads();
      if (b == 0) { rm[t] = new_mean; rv[t] = new_var; }
      ns[j] = (scores[j] - new_mean) * rsqrtf(new_var + NN_EPS) * node_gamma[t] + node_beta[t];
    }
    float mx = ns[0];
    for (int j = 1; j < ntar; ++j) mx = fmaxf(mx, ns[j]);
    float den = 0.f, ex[3];
    for (int j = 0; j < ntar; ++j) { ex[j] = __expf(ns[j] - mx); den += ex[j]; }
    float src = (nid == 0) ? 1.f : node_attns[nid*NB + b];
    for (int j = 0; j < ntar; ++j) attn_sent[j] = src * (ex[j] / den) * sbv;
  } else {
    attn_sent[0] = node_attns[nid*NB + b] * sbv;
  }

  float ad[8];
  for (int t = 0; t < 8; ++t) ad[t] = attn_dist[b*8 + t];
  if (nid > 0) ad[nid] = 0.f;
  for (int j = 0; j < ntar; ++j) ad[c_tar[nid][j]] += attn_sent[j];

  bool taken[8] = {false,false,false,false,false,false,false,false};
  float mask[8] = {0,0,0,0,0,0,0,0};
  float vsum = 0.f;
  for (int k = 0; k < 4; ++k) {
    float best = -__builtin_huge_valf(); int bi = 0;
    for (int t = 0; t < 8; ++t) if (!taken[t] && ad[t] > best) { best = ad[t]; bi = t; }
    taken[bi] = true; mask[bi] = 1.f;
    if (best > EPS_CUT) vsum += best;
  }
  for (int t = 0; t < 8; ++t) mask[t] *= (ad[t] > EPS_CUT) ? 1.f : 0.f;
  float scale = 1.f / vsum;
  for (int t = 0; t < 8; ++t) attn_dist[b*8 + t] = ad[t] * mask[t] * scale;

  if (nid == 0) {
    for (int j = 0; j < ntar; ++j) {
      int tid = c_tar[0][j];
      node_attns[tid*NB + b] = attn_sent[j] * mask[tid] * scale;
      tm[(tid*8 + 0)*NB + b] = mask[tid];
    }
  } else {
    for (int j = 0; j < ntar; ++j) {
      int tid = c_tar[nid][j];
      if (c_first[nid][j]) node_attns[tid*NB + b]  = attn_sent[j];
      else                 node_attns[tid*NB + b] += attn_sent[j];
      tm[(tid*8 + nid)*NB + b] = 1.f;
    }
    for (int p = 0; p < c_npart[nid]; ++p) {
      int tid = c_part[nid][p];
      float mcol = mask[tid];
      node_attns[tid*NB + b] *= mcol * scale;
      for (int si = 0; si < c_nsrc[tid]; ++si)
        tm[(tid*8 + c_src[tid][si])*NB + b] *= mcol;
    }
  }
}

// ---------------- per-source coefficients + subnet mask for node nid ----------------
__global__ void k_coef(int nid, const float* node_attns, float* sb, const float* tm,
                       float* coefbuf) {
  int b = threadIdx.x;
  int ns = c_nsrc[nid];
  float any = 0.f;
  for (int s = 0; s < ns; ++s) {
    int sid = c_src[nid][s];
    float eff = tm[(nid*8 + sid)*NB + b] * sb[sid*NB + b];
    coefbuf[s*NB + b] = (sid == 0) ? eff : eff * node_attns[sid*NB + b];
    any += eff;
  }
  sb[nid*NB + b] = (any > 0.f) ? 1.f : 0.f;
}

// ---------------- weighted accumulation -> f16 padded-NHWC conv input ----------------
__global__ void k_accum(const float* __restrict__ s0, const float* __restrict__ s1,
                        const float* __restrict__ s2, int ns,
                        const float* __restrict__ coefbuf, _Float16* __restrict__ acc) {
  size_t i = (size_t)blockIdx.x * blockDim.x + threadIdx.x;     // NCHW linear
  const size_t total = (size_t)NB * NCH * NH * NW;
  if (i >= total) return;
  size_t b   = i / ((size_t)NCH*NH*NW);
  size_t rem = i % ((size_t)NCH*NH*NW);
  size_t c   = rem / (NH*NW);
  size_t hw  = rem % (NH*NW);
  size_t h   = hw / NW, w = hw % NW;
  float v = coefbuf[b] * s0[i];
  if (ns > 1) v += coefbuf[NB + b]   * s1[i];
  if (ns > 2) v += coefbuf[2*NB + b] * s2[i];
  acc[((b*PH + h + 1)*PW + (w + 1))*NCH + c] = (_Float16)v;     // padded NHWC
}

// ---------------- node 7: fp32 weighted sum of nodes 5,6 -> d_out ----------------
__global__ void k_accum_out(const float* __restrict__ s5, const float* __restrict__ s6,
                            const float* __restrict__ coefbuf, float* __restrict__ out) {
  size_t i = (size_t)blockIdx.x * blockDim.x + threadIdx.x;
  const size_t total = (size_t)NB * 256 * 14 * 14;
  if (i >= total) return;
  size_t b = i / ((size_t)256*14*14);
  out[i] = coefbuf[b] * s5[i] + coefbuf[NB + b] * s6[i];
}

// ---------------- implicit-GEMM conv via v_wmma_f32_16x16x32_f16 ----------------
// X: f16 padded NHWC [B,30,30,128] (zero halo); Wt: f16 [Co][k], k=(kh*3+kw)*128+ci
// Out: fp32 NCHW, bias+ReLU fused. Block = 8 waves sharing one 64-wide N tile:
// the 64pos x 128ci x f16 input slice (16KB) is staged in LDS (async DMA when
// available) once per 3x3 tap and consumed by all 8 waves. Wave = 16(M) x 64(N).
__global__ void __launch_bounds__(256)
k_conv(const _Float16* __restrict__ X, const _Float16* __restrict__ Wt,
       const float* __restrict__ bias, float* __restrict__ Out,
       int Co, int Ho, int Wo, int stride, int pad, int MT)
{
  __shared__ __align__(16) _Float16 tile[64 * NCH];   // 16KB
  int tid  = threadIdx.x;
  int wid  = blockIdx.x * 8 + (tid >> 5);
  int lane = tid & 31;
  int mt = wid % MT, nt = wid / MT;
  int hs = (lane >> 4) & 1;
  int lr = lane & 15;

  // staging assignment: thread covers position p, 64B chunk q
  int p  = tid >> 2;
  int q  = tid & 3;
  int np  = nt*64 + p;
  int pb  = np / (Ho*Wo);
  int pr  = np % (Ho*Wo);
  int poh = pr / Wo, pow_ = pr % Wo;
  int base_h = poh*stride - pad + 1;     // +kh -> [0,PH)
  int base_w = pow_*stride - pad + 1;    // +kw -> [0,PW)
  _Float16* lrow = &tile[p*NCH + q*32];

  // epilogue positions for this lane's 4 fragments
  int enb[4], eoh[4], eow[4];
#pragma unroll
  for (int f = 0; f < 4; ++f) {
    int n = nt*64 + f*16 + lr;
    enb[f] = n / (Ho*Wo);
    int r2 = n % (Ho*Wo);
    eoh[f] = r2 / Wo; eow[f] = r2 % Wo;
  }

  v8f acc[4];
  const v8f z8 = {0.f,0.f,0.f,0.f,0.f,0.f,0.f,0.f};
#pragma unroll
  for (int f = 0; f < 4; ++f) acc[f] = z8;

  const _Float16* wbase = Wt + (size_t)(mt*16 + lr)*1152 + hs*8;

  for (int r = 0; r < 9; ++r) {
    int kh = r / 3, kw = r % 3;
    const _Float16* grow =
        X + ((((size_t)pb*PH) + base_h + kh)*PW + base_w + kw)*NCH + q*32;

    __syncthreads();   // previous tile fully consumed
#if HAVE_ASYNC_LDS
#pragma unroll
    for (int j = 0; j < 4; ++j)
      __builtin_amdgcn_global_load_async_to_lds_b128(
          (__attribute__((address_space(1))) i32x4*)(grow + j*8),
          (__attribute__((address_space(3))) i32x4*)(lrow + j*8), 0, 0);
#if __has_builtin(__builtin_amdgcn_s_wait_asynccnt)
    __builtin_amdgcn_s_wait_asynccnt(0);
#else
    asm volatile("s_wait_asynccnt 0" ::: "memory");
#endif
#else
    f32x4 t0 = *(const f32x4*)(grow);
    f32x4 t1 = *(const f32x4*)(grow + 8);
    f32x4 t2 = *(const f32x4*)(grow + 16);
    f32x4 t3 = *(const f32x4*)(grow + 24);
    *(f32x4*)(lrow)      = t0;
    *(f32x4*)(lrow + 8)  = t1;
    *(f32x4*)(lrow + 16) = t2;
    *(f32x4*)(lrow + 24) = t3;
#endif
    __syncthreads();   // tile visible to all waves

#pragma unroll
    for (int cb = 0; cb < 4; ++cb) {
      int kstep = r*4 + cb;
      union { v16h v; f32x4 qv[2]; } A;
      const _Float16* wk = wbase + kstep*32;
      A.qv[0] = *(const f32x4*)(wk);
      A.qv[1] = *(const f32x4*)(wk + 16);
      int cio = cb*32 + hs*8;
#pragma unroll
      for (int f = 0; f < 4; ++f) {
        const _Float16* bp = &tile[(f*16 + lr)*NCH + cio];
        union { v16h v; f32x4 qv[2]; } Bv;
        Bv.qv[0] = *(const f32x4*)(bp);
        Bv.qv[1] = *(const f32x4*)(bp + 16);
        acc[f] = __builtin_amdgcn_wmma_f32_16x16x32_f16(
            false, A.v, false, Bv.v, (short)0, acc[f], false, false);
      }
    }
  }

  // epilogue: bias + ReLU, scatter to NCHW
#pragma unroll
  for (int f = 0; f < 4; ++f) {
    size_t obase = ((size_t)enb[f]*Co)*(Ho*Wo) + (size_t)eoh[f]*Wo + eow[f];
#pragma unroll
    for (int r8 = 0; r8 < 8; ++r8) {
      int co = mt*16 + hs*8 + r8;
      float v = acc[f][r8] + bias[co];
      Out[obase + (size_t)co*(Ho*Wo)] = fmaxf(v, 0.f);
    }
  }
}

// ---------------- host orchestration ----------------
static const int h_nsrc[8]   = {0,1,2,2,3,3,2,2};
static const int h_src[8][3] = {{0,0,0},{0,0,0},{0,1,0},{0,1,0},{1,2,3},{2,3,4},{3,4,0},{5,6,0}};

extern "C" void kernel_launch(void* const* d_in, const int* in_sizes, int n_in,
                              void* d_out, int out_size, void* d_ws, size_t ws_size,
                              hipStream_t stream) {
  const float* x          = (const float*)d_in[0];
  const float* node_embs  = (const float*)d_in[1];
  const float* node_gamma = (const float*)d_in[2];
  const float* node_beta  = (const float*)d_in[3];
  const float* attend_w   = (const float*)d_in[4];
  const float* w_mid      = (const float*)d_in[5];
  const float* b_mid      = (const float*)d_in[6];
  const float* w_out      = (const float*)d_in[7];
  const float* b_out      = (const float*)d_in[8];
  const float* run_mean   = (const float*)d_in[9];
  const float* run_var    = (const float*)d_in[10];

  char* ws = (char*)d_ws;
  size_t off = 0;
  auto alloc = [&](size_t bytes) -> void* {
    void* p = ws + off;
    off = (off + bytes + 255) & ~(size_t)255;
    return p;
  };

  const size_t MIDE = (size_t)NB * NCH * NH * NW;         // 12.85M elems
  const size_t OUTE = (size_t)NB * 256 * 14 * 14;         // 6.42M elems
  const size_t PADE = (size_t)NB * PH * PW * NCH;         // padded f16 elems

  float* node[8];
  node[0] = (float*)x;
  for (int i = 1; i <= 4; ++i) node[i] = (float*)alloc(MIDE * 4);
  node[5] = (float*)alloc(OUTE * 4);
  node[6] = (float*)alloc(OUTE * 4);
  _Float16* accf   = (_Float16*)alloc(PADE * 2);
  _Float16* wt_mid = (_Float16*)alloc((size_t)4 * 128 * 1152 * 2);
  _Float16* wt_out = (_Float16*)alloc((size_t)2 * 256 * 1152 * 2);
  float* pooled     = (float*)alloc((size_t)NB * 256 * 4);
  float* attn_dist  = (float*)alloc(NB * 8 * 4);
  float* node_attns = (float*)alloc(8 * NB * 4);
  float* sbm        = (float*)alloc(8 * NB * 4);
  float* tmb        = (float*)alloc(64 * NB * 4);
  float* rm         = (float*)alloc(256);
  float* rv         = (float*)alloc(256);
  float* coefb      = (float*)alloc(3 * NB * 4);

  k_init<<<1, 256, 0, stream>>>(attn_dist, node_attns, sbm, tmb, rm, rv, run_mean, run_var);
  k_zerop<<<(int)((PADE/8 + 255)/256), 256, 0, stream>>>(accf, PADE/8);

  for (int c = 0; c < 4; ++c)
    k_wconv<<<(128*1152 + 255)/256, 256, 0, stream>>>(w_mid + (size_t)c*128*128*9,
                                                      wt_mid + (size_t)c*128*1152, 128);
  for (int c = 0; c < 2; ++c)
    k_wconv<<<(256*1152 + 255)/256, 256, 0, stream>>>(w_out + (size_t)c*256*128*9,
                                                      wt_out + (size_t)c*256*1152, 256);

  // stage 0
  k_pool<<<(NB*128)/8, 256, 0, stream>>>(x, pooled, 128, NH*NW);
  k_attn<<<1, NB, 0, stream>>>(0, pooled, attend_w, node_embs, node_gamma, node_beta,
                               rm, rv, attn_dist, node_attns, sbm, tmb);

  for (int nid = 1; nid <= 6; ++nid) {
    k_coef<<<1, NB, 0, stream>>>(nid, node_attns, sbm, tmb, coefb);
    const float* s0 = node[h_src[nid][0]];
    const float* s1 = node[h_src[nid][1]];
    const float* s2 = node[h_src[nid][2]];
    k_accum<<<(int)((MIDE + 255)/256), 256, 0, stream>>>(s0, s1, s2, h_nsrc[nid], coefb, accf);

    if (nid <= 4) {
      // 128ch out, 28x28, stride 1, pad 1 : MT=8, N=100352 -> 1568 blocks
      k_conv<<<1568, 256, 0, stream>>>(accf, wt_mid + (size_t)(nid-1)*128*1152,
                                       b_mid + (nid-1)*128, node[nid],
                                       128, 28, 28, 1, 1, 8);
      k_pool<<<(NB*128)/8, 256, 0, stream>>>(node[nid], pooled, 128, NH*NW);
    } else {
      // 256ch out, 14x14, stride 2, pad_lo 0 (JAX SAME asymmetric): MT=16 -> 784 blocks
      k_conv<<<784, 256, 0, stream>>>(accf, wt_out + (size_t)(nid-5)*256*1152,
                                      b_out + (nid-5)*256, node[nid],
                                      256, 14, 14, 2, 0, 16);
    }
    k_attn<<<1, NB, 0, stream>>>(nid, pooled, attend_w, node_embs, node_gamma, node_beta,
                                 rm, rv, attn_dist, node_attns, sbm, tmb);
  }

  // node 7: final weighted sum -> d_out (fp32 NCHW)
  k_coef<<<1, NB, 0, stream>>>(7, node_attns, sbm, tmb, coefb);
  k_accum_out<<<(int)((OUTE + 255)/256), 256, 0, stream>>>(node[5], node[6], coefb, (float*)d_out);
}
